// DensePermutation_16423954940015
// MI455X (gfx1250) — matrix operand order repaired and checked by reading, not compiled, not run on hardware
//
#include <hip/hip_runtime.h>

typedef __attribute__((ext_vector_type(2)))  float  v2f;
typedef __attribute__((ext_vector_type(4)))  float  v4f;
typedef __attribute__((ext_vector_type(8)))  float  v8f;
typedef __attribute__((ext_vector_type(8)))  __bf16 v8bf;
typedef __attribute__((ext_vector_type(16))) __bf16 v16bf;

#define N1 8192   // feat1 rows  (softmax axis, columns of P21)
#define N2 8192   // feat2 rows  (rows of P21)
#define DIM 128
#define INV_BLUR 10.0f

// ---------------------------------------------------------------------------
// Prologue 1: row squared-norms (exact, f32)
// ---------------------------------------------------------------------------
__global__ void rownorm_kernel(const float* __restrict__ f, float* __restrict__ sq) {
  const int r = blockIdx.x * blockDim.x + threadIdx.x;
  const float* p = f + (size_t)r * DIM;
  float s = 0.f;
#pragma unroll
  for (int k = 0; k < DIM; k += 4) {
    v4f v = *(const v4f*)(p + k);
    s = fmaf(v.x, v.x, s); s = fmaf(v.y, v.y, s);
    s = fmaf(v.z, v.z, s); s = fmaf(v.w, v.w, s);
  }
  sq[r] = s;
}

// ---------------------------------------------------------------------------
// Prologue 2: split f32 -> bf16 hi + bf16 lo  (x ~= hi + lo, ~16 mantissa bits)
// ---------------------------------------------------------------------------
__global__ void split_bf16_kernel(const float* __restrict__ src,
                                  __bf16* __restrict__ hi, __bf16* __restrict__ lo) {
  const int i = blockIdx.x * blockDim.x + threadIdx.x;
  const float x = src[i];
  const __bf16 h = (__bf16)x;
  hi[i] = h;
  lo[i] = (__bf16)(x - (float)h);
}

// ---------------------------------------------------------------------------
// Fragment loaders for V_WMMA_F32_16X16X32_BF16 (ISA 7.12.2, 16-bit layouts)
// A 16x32: lane l<16 holds M=l, K={0..7,16..23}; lanes 16-31 K={8..15,24..31}
// B 32x16: lane l<16 holds N=l, K={0..7,16..23}; lanes 16-31 K={8..15,24..31}
// ---------------------------------------------------------------------------
__device__ __forceinline__ v16bf load_a_frag(const __bf16* base, int hb) {
  v8bf p0 = *(const v8bf*)(base + hb * 8);         // K = kc*32 + 8*hb + 0..7
  v8bf p1 = *(const v8bf*)(base + 16 + hb * 8);    // K = kc*32 + 16 + 8*hb + 0..7
  return __builtin_shufflevector(p0, p1, 0,1,2,3,4,5,6,7,8,9,10,11,12,13,14,15);
}
__device__ __forceinline__ v16bf load_b_frag(const __bf16* base, int hb) {
  v8bf p0 = *(const v8bf*)(base + hb * 8);         // K = kc*32 + 8*hb + 0..7
  v8bf p1 = *(const v8bf*)(base + 16 + hb * 8);    // K = kc*32 + 16 + 8*hb + 0..7
  return __builtin_shufflevector(p0, p1, 0,1,2,3,4,5,6,7,8,9,10,11,12,13,14,15);
}

#define WMMA_BF16(A, B, C) \
  __builtin_amdgcn_wmma_f32_16x16x32_bf16(false, (A), false, (B), (short)0, (C), false, false)

// Load the full B fragment set (hi+lo, K=128) for one 16-column tile.
__device__ __forceinline__ void load_tileB(const __bf16* __restrict__ f1hi,
                                           const __bf16* __restrict__ f1lo,
                                           int col, int hb,
                                           v16bf* __restrict__ BH,
                                           v16bf* __restrict__ BL) {
  const __bf16* bch = f1hi + (size_t)col * DIM;
  const __bf16* bcl = f1lo + (size_t)col * DIM;
#pragma unroll
  for (int kc = 0; kc < 4; ++kc) {
    BH[kc] = load_b_frag(bch + kc * 32, hb);
    BL[kc] = load_b_frag(bcl + kc * 32, hb);
  }
}

// bf16x3 GEMM for one 16x16 tile: cross ~ (Ahi+Alo)(Bhi+Blo), drop lo*lo.
__device__ __forceinline__ v8f compute_tile(const v16bf* __restrict__ AH,
                                            const v16bf* __restrict__ AL,
                                            const v16bf* __restrict__ BH,
                                            const v16bf* __restrict__ BL) {
  v8f c = {};
#pragma unroll
  for (int kc = 0; kc < 4; ++kc) {
    c = WMMA_BF16(AH[kc], BH[kc], c);
    c = WMMA_BF16(AL[kc], BH[kc], c);
    c = WMMA_BF16(AH[kc], BL[kc], c);
  }
  return c;
}

// ---------------------------------------------------------------------------
// Main kernel (bf16x3 path). One block = 16 output rows; 8 waves split the
// 8192 columns into 1024-wide strips. Double-buffered B fragments so the XDL
// pipe never waits on loadcnt==0, and tile-n epilogue VALU co-executes with
// tile-(n+1) WMMAs.
//  pass 1: GEMM -> per-row min squared distance
//  pass 2: GEMM -> exp((dmin-d)/blur), row sums, store unnormalized (L2-hot)
//  pass 3: re-read strip from L2, scale by 1/rowsum, non-temporal store
// ---------------------------------------------------------------------------
__global__ __launch_bounds__(256, 1)
void DensePermutation_bf16x3_kernel(const __bf16* __restrict__ f1hi,
                                    const __bf16* __restrict__ f1lo,
                                    const __bf16* __restrict__ f2hi,
                                    const __bf16* __restrict__ f2lo,
                                    const float*  __restrict__ sq1,
                                    const float*  __restrict__ sq2,
                                    float* __restrict__ out) {
  const int tid  = threadIdx.x;
  const int w    = tid >> 5;
  const int lane = tid & 31;
  const int hb   = lane >> 4;
  const int l16  = lane & 15;
  const int row_base = blockIdx.x * 16;

  __shared__ float s_red[8][16];

  // A fragments (16 rows x K=128, hi+lo), register resident: 64 VGPRs
  v16bf AH[4], AL[4];
  {
    const __bf16* arh = f2hi + (size_t)(row_base + l16) * DIM;
    const __bf16* arl = f2lo + (size_t)(row_base + l16) * DIM;
#pragma unroll
    for (int kc = 0; kc < 4; ++kc) {
      AH[kc] = load_a_frag(arh + kc * 32, hb);
      AL[kc] = load_a_frag(arl + kc * 32, hb);
    }
  }
  // ||feat2[m]||^2 re-indexed to C layout (m = g + 8*hb)
  float rsq2[8];
#pragma unroll
  for (int g = 0; g < 8; ++g) rsq2[g] = sq2[row_base + g + 8 * hb];

  const int n_begin = w * (N1 / 8);
  const int n_end   = n_begin + (N1 / 8);

  v16bf BH0[4], BL0[4], BH1[4], BL1[4];

  // ---------------- pass 1: min squared distance ----------------
  float sqmin[8];
#pragma unroll
  for (int g = 0; g < 8; ++g) sqmin[g] = 3.4e38f;

  load_tileB(f1hi, f1lo, n_begin + l16, hb, BH0, BL0);
  for (int n0 = n_begin; n0 < n_end; n0 += 32) {   // strip is 64 tiles: even
    load_tileB(f1hi, f1lo, n0 + 16 + l16, hb, BH1, BL1);
    const float s1a = sq1[n0 + l16];
    const float s1b = sq1[n0 + 16 + l16];
    v8f c0 = compute_tile(AH, AL, BH0, BL0);
    if (n0 + 32 < n_end) load_tileB(f1hi, f1lo, n0 + 32 + l16, hb, BH0, BL0);
    v8f c1 = compute_tile(AH, AL, BH1, BL1);
    // c0 epilogue overlaps c1's WMMA chain
#pragma unroll
    for (int g = 0; g < 8; ++g)
      sqmin[g] = fminf(sqmin[g], fmaf(-2.f, c0[g], rsq2[g] + s1a));
#pragma unroll
    for (int g = 0; g < 8; ++g)
      sqmin[g] = fminf(sqmin[g], fmaf(-2.f, c1[g], rsq2[g] + s1b));
  }
#pragma unroll
  for (int g = 0; g < 8; ++g) {
#pragma unroll
    for (int m = 1; m < 16; m <<= 1)
      sqmin[g] = fminf(sqmin[g], __shfl_xor(sqmin[g], m));
  }
  if (l16 == 0) {
#pragma unroll
    for (int g = 0; g < 8; ++g) s_red[w][g + 8 * hb] = sqmin[g];
  }
  __syncthreads();
  float rowmin[8];   // min distance per row (after sqrt)
#pragma unroll
  for (int g = 0; g < 8; ++g) {
    const int m = g + 8 * hb;
    float v = s_red[0][m];
#pragma unroll
    for (int ww = 1; ww < 8; ++ww) v = fminf(v, s_red[ww][m]);
    rowmin[g] = sqrtf(fmaxf(v, 0.f));
  }
  __syncthreads();

  // ------- pass 2: exp + row sums, store unnormalized (stays in L2) -------
  float rsum[8];
#pragma unroll
  for (int g = 0; g < 8; ++g) rsum[g] = 0.f;

  load_tileB(f1hi, f1lo, n_begin + l16, hb, BH0, BL0);
  for (int n0 = n_begin; n0 < n_end; n0 += 32) {
    load_tileB(f1hi, f1lo, n0 + 16 + l16, hb, BH1, BL1);
    const float s1a = sq1[n0 + l16];
    const float s1b = sq1[n0 + 16 + l16];
    v8f c0 = compute_tile(AH, AL, BH0, BL0);
    if (n0 + 32 < n_end) load_tileB(f1hi, f1lo, n0 + 32 + l16, hb, BH0, BL0);
    v8f c1 = compute_tile(AH, AL, BH1, BL1);
#pragma unroll
    for (int g = 0; g < 8; ++g) {
      float sq = fmaf(-2.f, c0[g], rsq2[g] + s1a);
      float d  = sqrtf(fmaxf(sq, 0.f));
      float e  = __expf((rowmin[g] - d) * INV_BLUR);
      rsum[g] += e;
      out[(size_t)(row_base + g + 8 * hb) * N1 + (n0 + l16)] = e;
    }
#pragma unroll
    for (int g = 0; g < 8; ++g) {
      float sq = fmaf(-2.f, c1[g], rsq2[g] + s1b);
      float d  = sqrtf(fmaxf(sq, 0.f));
      float e  = __expf((rowmin[g] - d) * INV_BLUR);
      rsum[g] += e;
      out[(size_t)(row_base + g + 8 * hb) * N1 + (n0 + 16 + l16)] = e;
    }
  }
#pragma unroll
  for (int g = 0; g < 8; ++g) {
#pragma unroll
    for (int m = 1; m < 16; m <<= 1)
      rsum[g] += __shfl_xor(rsum[g], m);
  }
  if (l16 == 0) {
#pragma unroll
    for (int g = 0; g < 8; ++g) s_red[w][g + 8 * hb] = rsum[g];
  }
  __syncthreads();
  float inv[16];
#pragma unroll
  for (int m = 0; m < 16; ++m) {
    float v = 0.f;
#pragma unroll
    for (int ww = 0; ww < 8; ++ww) v += s_red[ww][m];
    inv[m] = 1.0f / v;
  }

  asm volatile("s_wait_storecnt 0x0" ::: "memory");  // our streamed stores done
  __syncthreads();

  // ---------------- pass 3: normalize (L2 read, NT write) ----------------
#pragma unroll
  for (int m = 0; m < 16; ++m) {
    float* rowp = out + (size_t)(row_base + m) * N1;
    const float im = inv[m];
    for (int nc = n_begin + lane * 4; nc < n_end; nc += 128) {
      v4f v = *(const v4f*)(rowp + nc);
      v.x *= im; v.y *= im; v.z *= im; v.w *= im;
      __builtin_nontemporal_store(v, (v4f*)(rowp + nc));
    }
  }
}

// ---------------------------------------------------------------------------
// Fallback (no workspace): self-contained f32 WMMA version
// ---------------------------------------------------------------------------
__global__ __launch_bounds__(256, 1)
void DensePermutation_f32_kernel(const float* __restrict__ feat1,
                                 const float* __restrict__ feat2,
                                 float* __restrict__ out) {
  const int tid  = threadIdx.x;
  const int w    = tid >> 5;
  const int lane = tid & 31;
  const int hb   = lane >> 4;
  const int l16  = lane & 15;
  const int row_base = blockIdx.x * 16;

  __shared__ float s_red[8][16];

  v2f a[32];
  const float* arow = feat2 + (size_t)(row_base + l16) * DIM + 2 * hb;
  float s2 = 0.f;
#pragma unroll
  for (int t = 0; t < 32; ++t) {
    a[t] = *(const v2f*)(arow + 4 * t);
    s2 = fmaf(a[t].x, a[t].x, s2);
    s2 = fmaf(a[t].y, a[t].y, s2);
  }
  s2 += __shfl_xor(s2, 16);
  float rsq2[8];
#pragma unroll
  for (int g = 0; g < 8; ++g) rsq2[g] = __shfl(s2, g + 8 * hb);

  const int n_begin = w * (N1 / 8);
  const int n_end   = n_begin + (N1 / 8);

  float sqmin[8];
#pragma unroll
  for (int g = 0; g < 8; ++g) sqmin[g] = 3.4e38f;

  for (int n0 = n_begin; n0 < n_end; n0 += 16) {
    const float* brow = feat1 + (size_t)(n0 + l16) * DIM + 2 * hb;
    v8f c = {};
    float s1 = 0.f;
#pragma unroll
    for (int t = 0; t < 32; ++t) {
      v2f b = *(const v2f*)(brow + 4 * t);
      s1 = fmaf(b.x, b.x, s1);
      s1 = fmaf(b.y, b.y, s1);
      c = __builtin_amdgcn_wmma_f32_16x16x4_f32(false, a[t], false, b,
                                                (short)0, c, false, false);
    }
    s1 += __shfl_xor(s1, 16);
#pragma unroll
    for (int g = 0; g < 8; ++g)
      sqmin[g] = fminf(sqmin[g], fmaf(-2.f, c[g], rsq2[g] + s1));
  }
#pragma unroll
  for (int g = 0; g < 8; ++g) {
#pragma unroll
    for (int m = 1; m < 16; m <<= 1)
      sqmin[g] = fminf(sqmin[g], __shfl_xor(sqmin[g], m));
  }
  if (l16 == 0) {
#pragma unroll
    for (int g = 0; g < 8; ++g) s_red[w][g + 8 * hb] = sqmin[g];
  }
  __syncthreads();
  float rowmin[8];
#pragma unroll
  for (int g = 0; g < 8; ++g) {
    const int m = g + 8 * hb;
    float v = s_red[0][m];
#pragma unroll
    for (int ww = 1; ww < 8; ++ww) v = fminf(v, s_red[ww][m]);
    rowmin[g] = sqrtf(fmaxf(v, 0.f));
  }
  __syncthreads();

  float rsum[8];
#pragma unroll
  for (int g = 0; g < 8; ++g) rsum[g] = 0.f;

  for (int n0 = n_begin; n0 < n_end; n0 += 16) {
    const float* brow = feat1 + (size_t)(n0 + l16) * DIM + 2 * hb;
    v8f c = {};
    float s1 = 0.f;
#pragma unroll
    for (int t = 0; t < 32; ++t) {
      v2f b = *(const v2f*)(brow + 4 * t);
      s1 = fmaf(b.x, b.x, s1);
      s1 = fmaf(b.y, b.y, s1);
      c = __builtin_amdgcn_wmma_f32_16x16x4_f32(false, a[t], false, b,
                                                (short)0, c, false, false);
    }
    s1 += __shfl_xor(s1, 16);
#pragma unroll
    for (int g = 0; g < 8; ++g) {
      float sq = fmaf(-2.f, c[g], rsq2[g] + s1);
      float d  = sqrtf(fmaxf(sq, 0.f));
      float e  = __expf((rowmin[g] - d) * INV_BLUR);
      rsum[g] += e;
      out[(size_t)(row_base + g + 8 * hb) * N1 + (n0 + l16)] = e;
    }
  }
#pragma unroll
  for (int g = 0; g < 8; ++g) {
#pragma unroll
    for (int m = 1; m < 16; m <<= 1)
      rsum[g] += __shfl_xor(rsum[g], m);
  }
  if (l16 == 0) {
#pragma unroll
    for (int g = 0; g < 8; ++g) s_red[w][g + 8 * hb] = rsum[g];
  }
  __syncthreads();
  float inv[16];
#pragma unroll
  for (int m = 0; m < 16; ++m) {
    float v = 0.f;
#pragma unroll
    for (int ww = 0; ww < 8; ++ww) v += s_red[ww][m];
    inv[m] = 1.0f / v;
  }

  asm volatile("s_wait_storecnt 0x0" ::: "memory");
  __syncthreads();

#pragma unroll
  for (int m = 0; m < 16; ++m) {
    float* rowp = out + (size_t)(row_base + m) * N1;
    const float im = inv[m];
    for (int nc = n_begin + lane * 4; nc < n_end; nc += 128) {
      v4f v = *(const v4f*)(rowp + nc);
      v.x *= im; v.y *= im; v.z *= im; v.w *= im;
      __builtin_nontemporal_store(v, (v4f*)(rowp + nc));
    }
  }
}

extern "C" void kernel_launch(void* const* d_in, const int* in_sizes, int n_in,
                              void* d_out, int out_size, void* d_ws, size_t ws_size,
                              hipStream_t stream) {
  const float* feat1 = (const float*)d_in[0];  // [8192, 128]
  const float* feat2 = (const float*)d_in[1];  // [8192, 128]
  float* out = (float*)d_out;                  // [8192, 8192]
  (void)in_sizes; (void)n_in; (void)out_size;

  const size_t nElem = (size_t)N1 * DIM;                       // 1M per matrix
  const size_t need  = 2 * 32768 + 4 * nElem * sizeof(__bf16); // 64KB + 8MB

  if (ws_size >= need) {
    char*   ws   = (char*)d_ws;
    float*  sq1  = (float*)ws;
    float*  sq2  = (float*)(ws + 32768);
    __bf16* f1hi = (__bf16*)(ws + 65536);
    __bf16* f1lo = f1hi + nElem;
    __bf16* f2hi = f1lo + nElem;
    __bf16* f2lo = f2hi + nElem;

    rownorm_kernel<<<N1 / 256, 256, 0, stream>>>(feat1, sq1);
    rownorm_kernel<<<N2 / 256, 256, 0, stream>>>(feat2, sq2);
    split_bf16_kernel<<<(int)(nElem / 256), 256, 0, stream>>>(feat1, f1hi, f1lo);
    split_bf16_kernel<<<(int)(nElem / 256), 256, 0, stream>>>(feat2, f2hi, f2lo);

    DensePermutation_bf16x3_kernel<<<N2 / 16, 256, 0, stream>>>(
        f1hi, f1lo, f2hi, f2lo, sq1, sq2, out);
  } else {
    DensePermutation_f32_kernel<<<N2 / 16, 256, 0, stream>>>(feat1, feat2, out);
  }
}
